// Seq2Seq_8014408975023
// MI455X (gfx1250) — matrix operand order repaired and checked by reading, not compiled, not run on hardware
//
#include <hip/hip_runtime.h>
#include <hip/hip_bf16.h>

// Seq2Seq LSTM (enc 2x256 steps, dec 2x30 steps + FC) for MI455X / gfx1250.
// bf16 WMMA (v_wmma_f32_16x16x32_bf16) with f32 accumulation for all GEMMs;
// h state kept in bf16 (GEMM input only), c state in f32. Weights L2-resident;
// recurrence is latency-bound -> one fused step kernel per (layer, timestep),
// replayed via graph capture. Each wave computes a 32(M)x16(N) region for all
// four gates (8 accumulators), reusing each W_hh fragment for 2 WMMAs
// (1.5 loads/WMMA), and does the cell nonlinearity fully in-register.

#define B_     512
#define T_ENC_ 256
#define IN_    32
#define HID_   256
#define G4_    1024
#define PW_    30

typedef __attribute__((ext_vector_type(16))) __bf16 v16bf;
typedef __attribute__((ext_vector_type(8)))  float  v8f;

union Frag16 { v16bf v; uint4 q[2]; };

__device__ __forceinline__ float sigmoidf_(float x) {
    return 1.0f / (1.0f + __expf(-x));
}
__device__ __forceinline__ float tanh_fast(float x) {
    x = fminf(fmaxf(x, -15.0f), 15.0f);
    float e = __expf(2.0f * x);
    return (e - 1.0f) / (e + 1.0f);
}

// ---------------- utility kernels ----------------

__global__ void cvt_f32_bf16(const float* __restrict__ in, __bf16* __restrict__ out, int n) {
    for (int i = blockIdx.x * blockDim.x + threadIdx.x; i < n; i += gridDim.x * blockDim.x)
        out[i] = (__bf16)in[i];
}

__global__ void bias_combine(const float* __restrict__ a, const float* __restrict__ b,
                             float* __restrict__ o, int n) {
    int i = blockIdx.x * blockDim.x + threadIdx.x;
    if (i < n) o[i] = a[i] + b[i];
}

__global__ void zero_u4(uint4* __restrict__ p, int n4) {
    int i = blockIdx.x * blockDim.x + threadIdx.x;
    if (i < n4) { uint4 z; z.x = z.y = z.z = z.w = 0u; p[i] = z; }
}

// ---------------- fused LSTM step ----------------
// gates[b, g*H+j] = bias[g*H+j] + sum_k h[b,k]*Whh[g*H+j, k]
//                 (+ sum_k x[b,k]*Wx[g*H+j, k])            if kx_tiles > 0
//                 (+ xscal[b]*Wx1[g*H+j])                  if xscal != nullptr
// then c' = sig(f)*c + sig(i)*tanh(g);  h' = sig(o)*tanh(c')
// Grid: 32 blocks x 256 threads (8 waves). 256 wave jobs =
//   16 M-supertiles (32 rows) x 16 J-tiles (16 cols).
__global__ __launch_bounds__(256) void lstm_step(
    const __bf16* __restrict__ hin,          // [B, H] bf16
    const __bf16* __restrict__ Whh,          // [4H, H] bf16, row-major
    const __bf16* __restrict__ x, int xstride, int kx_tiles,
    const __bf16* __restrict__ Wx,           // [4H, Dx] bf16 (Dx = kx_tiles*32)
    const float*  __restrict__ xscal, int xscal_stride,  // scalar input path (Dx==1)
    const float*  __restrict__ Wx1,          // [4H] f32 (scalar path weight col)
    const float*  __restrict__ bias,         // [4H] f32 (b_ih + b_hh)
    float*        __restrict__ c,            // [B, H] f32, in-place
    __bf16*       __restrict__ hout)         // [B, H] bf16
{
    const int wave  = threadIdx.x >> 5;
    const int lane  = threadIdx.x & 31;
    const int job   = blockIdx.x * 8 + wave;     // 0..255
    const int m0    = (job >> 4) << 5;           // 16 supertiles of 32 batch rows
    const int j0    = (job & 15) << 4;           // 16 hidden-col tiles
    const int lhalf = lane >> 4;                 // 0/1
    const int l16   = lane & 15;

    v8f acc[2][4] = {};                          // [m-subtile][gate i,f,g,o]

    const int wrow = j0 + l16;                   // B-fragment col (within H)
    const __bf16* Ap0 = hin + (size_t)(m0 + l16) * HID_;
    const __bf16* Ap1 = hin + (size_t)(m0 + 16 + l16) * HID_;

    // Recurrent GEMM: h @ Whh^T, K = 256 (8 k-tiles of 32)
    #pragma unroll
    for (int kt = 0; kt < 8; ++kt) {
        Frag16 a0, a1;
        const int ka = kt * 32 + lhalf * 8;
        a0.q[0] = *(const uint4*)(Ap0 + ka);
        a0.q[1] = *(const uint4*)(Ap0 + ka + 16);
        a1.q[0] = *(const uint4*)(Ap1 + ka);
        a1.q[1] = *(const uint4*)(Ap1 + ka + 16);
        const int kb = kt * 32 + lhalf * 8;
        #pragma unroll
        for (int gg = 0; gg < 4; ++gg) {
            Frag16 b;
            const __bf16* pb = Whh + (size_t)(gg * HID_ + wrow) * HID_ + kb;
            b.q[0] = *(const uint4*)(pb);
            b.q[1] = *(const uint4*)(pb + 16);
            acc[0][gg] = __builtin_amdgcn_wmma_f32_16x16x32_bf16(
                false, a0.v, false, b.v, (short)0, acc[0][gg], false, false);
            acc[1][gg] = __builtin_amdgcn_wmma_f32_16x16x32_bf16(
                false, a1.v, false, b.v, (short)0, acc[1][gg], false, false);
        }
    }

    // Input GEMM: x @ Wx^T, K = kx_tiles*32 (uniform branch; EXEC untouched)
    if (kx_tiles > 0) {
        const int Dx = kx_tiles * 32;
        const __bf16* Xp0 = x + (size_t)(m0 + l16) * xstride;
        const __bf16* Xp1 = x + (size_t)(m0 + 16 + l16) * xstride;
        for (int kt = 0; kt < kx_tiles; ++kt) {
            Frag16 a0, a1;
            const int ka = kt * 32 + lhalf * 8;
            a0.q[0] = *(const uint4*)(Xp0 + ka);
            a0.q[1] = *(const uint4*)(Xp0 + ka + 16);
            a1.q[0] = *(const uint4*)(Xp1 + ka);
            a1.q[1] = *(const uint4*)(Xp1 + ka + 16);
            const int kb = kt * 32 + lhalf * 8;
            #pragma unroll
            for (int gg = 0; gg < 4; ++gg) {
                Frag16 b;
                const __bf16* pb = Wx + (size_t)(gg * HID_ + wrow) * Dx + kb;
                b.q[0] = *(const uint4*)(pb);
                b.q[1] = *(const uint4*)(pb + 16);
                acc[0][gg] = __builtin_amdgcn_wmma_f32_16x16x32_bf16(
                    false, a0.v, false, b.v, (short)0, acc[0][gg], false, false);
                acc[1][gg] = __builtin_amdgcn_wmma_f32_16x16x32_bf16(
                    false, a1.v, false, b.v, (short)0, acc[1][gg], false, false);
            }
        }
    }

    // Epilogue. C/D layout: elem v -> (b = m_base + lhalf*8 + v, j = j0 + l16)
    const int j  = j0 + l16;
    const float bi = bias[0 * HID_ + j];
    const float bf = bias[1 * HID_ + j];
    const float bg = bias[2 * HID_ + j];
    const float bo = bias[3 * HID_ + j];
    float wi = 0.f, wf = 0.f, wg = 0.f, wo = 0.f;
    if (xscal) {
        wi = Wx1[0 * HID_ + j]; wf = Wx1[1 * HID_ + j];
        wg = Wx1[2 * HID_ + j]; wo = Wx1[3 * HID_ + j];
    }
    #pragma unroll
    for (int u = 0; u < 2; ++u) {
        #pragma unroll
        for (int v = 0; v < 8; ++v) {
            const int brow = m0 + u * 16 + lhalf * 8 + v;
            float gi = acc[u][0][v] + bi;
            float gf = acc[u][1][v] + bf;
            float gc = acc[u][2][v] + bg;
            float go = acc[u][3][v] + bo;
            if (xscal) {
                const float xv = xscal[brow * xscal_stride];
                gi += xv * wi; gf += xv * wf; gc += xv * wg; go += xv * wo;
            }
            const float iv = sigmoidf_(gi);
            const float fv = sigmoidf_(gf);
            const float gv = tanh_fast(gc);
            const float ov = sigmoidf_(go);
            const int idx = brow * HID_ + j;
            const float cn = fv * c[idx] + iv * gv;
            c[idx] = cn;
            hout[idx] = (__bf16)(ov * tanh_fast(cn));
        }
    }
}

// ---------------- final FC (per decoder step) ----------------
__global__ void fc_step(const __bf16* __restrict__ h, const float* __restrict__ fcw,
                        const float* __restrict__ fcb, float* __restrict__ out, int t) {
    int b = blockIdx.x * blockDim.x + threadIdx.x;
    if (b >= B_) return;
    float s = fcb[0];
    const __bf16* hp = h + (size_t)b * HID_;
    union { uint4 q; __bf16 e[8]; } hv;
    for (int jj = 0; jj < HID_; jj += 8) {
        hv.q = *(const uint4*)(hp + jj);
        #pragma unroll
        for (int u = 0; u < 8; ++u) s += (float)hv.e[u] * fcw[jj + u];
    }
    out[b * PW_ + t] = s;
}

// ---------------- host launch ----------------

extern "C" void kernel_launch(void* const* d_in, const int* in_sizes, int n_in,
                              void* d_out, int out_size, void* d_ws, size_t ws_size,
                              hipStream_t stream) {
    (void)in_sizes; (void)n_in; (void)out_size; (void)ws_size;
    const float* input  = (const float*)d_in[0];
    const float* target = (const float*)d_in[1];
    const float* eWih0 = (const float*)d_in[2];
    const float* eWhh0 = (const float*)d_in[3];
    const float* eBih0 = (const float*)d_in[4];
    const float* eBhh0 = (const float*)d_in[5];
    const float* eWih1 = (const float*)d_in[6];
    const float* eWhh1 = (const float*)d_in[7];
    const float* eBih1 = (const float*)d_in[8];
    const float* eBhh1 = (const float*)d_in[9];
    const float* dWih0 = (const float*)d_in[10];
    const float* dWhh0 = (const float*)d_in[11];
    const float* dBih0 = (const float*)d_in[12];
    const float* dBhh0 = (const float*)d_in[13];
    const float* dWih1 = (const float*)d_in[14];
    const float* dWhh1 = (const float*)d_in[15];
    const float* dBih1 = (const float*)d_in[16];
    const float* dBhh1 = (const float*)d_in[17];
    const float* fcw   = (const float*)d_in[18];
    const float* fcb   = (const float*)d_in[19];
    float* out = (float*)d_out;

    // workspace carve-up (256B aligned)
    char* ws = (char*)d_ws;
    size_t off = 0;
    auto carve = [&](size_t bytes) -> void* {
        void* p = ws + off;
        off = (off + bytes + 255) & ~(size_t)255;
        return p;
    };
    __bf16* x_bf   = (__bf16*)carve((size_t)B_ * T_ENC_ * IN_ * 2);
    __bf16* eWih0b = (__bf16*)carve((size_t)G4_ * IN_  * 2);
    __bf16* eWhh0b = (__bf16*)carve((size_t)G4_ * HID_ * 2);
    __bf16* eWih1b = (__bf16*)carve((size_t)G4_ * HID_ * 2);
    __bf16* eWhh1b = (__bf16*)carve((size_t)G4_ * HID_ * 2);
    __bf16* dWhh0b = (__bf16*)carve((size_t)G4_ * HID_ * 2);
    __bf16* dWih1b = (__bf16*)carve((size_t)G4_ * HID_ * 2);
    __bf16* dWhh1b = (__bf16*)carve((size_t)G4_ * HID_ * 2);
    float* biasE0 = (float*)carve(G4_ * 4);
    float* biasE1 = (float*)carve(G4_ * 4);
    float* biasD0 = (float*)carve(G4_ * 4);
    float* biasD1 = (float*)carve(G4_ * 4);
    // state: h double-buffered bf16 [layer][parity][B][H], c f32 [layer][B][H]
    __bf16* hbuf = (__bf16*)carve((size_t)2 * 2 * B_ * HID_ * 2);   // 1 MiB
    float*  cbuf = (float*)carve((size_t)2 * B_ * HID_ * 4);        // 1 MiB
    auto hb = [&](int layer, int parity) -> __bf16* {
        return hbuf + (size_t)(layer * 2 + parity) * B_ * HID_;
    };

    // 1) convert inputs & weights to bf16, combine biases, zero state
    cvt_f32_bf16<<<1024, 256, 0, stream>>>(input, x_bf, B_ * T_ENC_ * IN_);
    cvt_f32_bf16<<<128, 256, 0, stream>>>(eWih0, eWih0b, G4_ * IN_);
    cvt_f32_bf16<<<512, 256, 0, stream>>>(eWhh0, eWhh0b, G4_ * HID_);
    cvt_f32_bf16<<<512, 256, 0, stream>>>(eWih1, eWih1b, G4_ * HID_);
    cvt_f32_bf16<<<512, 256, 0, stream>>>(eWhh1, eWhh1b, G4_ * HID_);
    cvt_f32_bf16<<<512, 256, 0, stream>>>(dWhh0, dWhh0b, G4_ * HID_);
    cvt_f32_bf16<<<512, 256, 0, stream>>>(dWih1, dWih1b, G4_ * HID_);
    cvt_f32_bf16<<<512, 256, 0, stream>>>(dWhh1, dWhh1b, G4_ * HID_);
    bias_combine<<<4, 256, 0, stream>>>(eBih0, eBhh0, biasE0, G4_);
    bias_combine<<<4, 256, 0, stream>>>(eBih1, eBhh1, biasE1, G4_);
    bias_combine<<<4, 256, 0, stream>>>(dBih0, dBhh0, biasD0, G4_);
    bias_combine<<<4, 256, 0, stream>>>(dBih1, dBhh1, biasD1, G4_);
    {   // zero h (all buffers) + c : contiguous 2 MiB
        size_t zbytes = (size_t)2 * 2 * B_ * HID_ * 2 + (size_t)2 * B_ * HID_ * 4;
        int n4 = (int)(zbytes / 16);
        zero_u4<<<(n4 + 255) / 256, 256, 0, stream>>>((uint4*)hbuf, n4);
    }

    const dim3 sgrid(32), sblk(256);

    // 2) encoder: 256 steps x 2 layers
    for (int t = 0; t < T_ENC_; ++t) {
        const int r = t & 1, w = r ^ 1;
        // layer 0: x_t from input [B,T,32] (row stride T*32), K_x = 32 (1 tile)
        lstm_step<<<sgrid, sblk, 0, stream>>>(
            hb(0, r), eWhh0b,
            x_bf + (size_t)t * IN_, T_ENC_ * IN_, 1, eWih0b,
            nullptr, 0, nullptr,
            biasE0, cbuf, hb(0, w));
        // layer 1: x = layer0's fresh h, K_x = 256 (8 tiles)
        lstm_step<<<sgrid, sblk, 0, stream>>>(
            hb(1, r), eWhh1b,
            hb(0, w), HID_, 8, eWih1b,
            nullptr, 0, nullptr,
            biasE1, cbuf + (size_t)B_ * HID_, hb(1, w));
    }

    // 3) decoder: 30 steps x 2 layers + FC (state continues from encoder)
    for (int t = 0; t < PW_; ++t) {
        const int s = T_ENC_ + t;
        const int r = s & 1, w = r ^ 1;
        // layer 0: scalar input target[:, t] (D=1) -> rank-1 VALU path
        lstm_step<<<sgrid, sblk, 0, stream>>>(
            hb(0, r), dWhh0b,
            nullptr, 0, 0, nullptr,
            target + t, PW_, dWih0,
            biasD0, cbuf, hb(0, w));
        // layer 1
        lstm_step<<<sgrid, sblk, 0, stream>>>(
            hb(1, r), dWhh1b,
            hb(0, w), HID_, 8, dWih1b,
            nullptr, 0, nullptr,
            biasD1, cbuf + (size_t)B_ * HID_, hb(1, w));
        // FC: out[:, t] = h1 @ fc_w^T + fc_b
        fc_step<<<2, 256, 0, stream>>>(hb(1, w), fcw, fcb, out, t);
    }
}
